// gru4recF_attention_44263932952655
// MI455X (gfx1250) — compile-verified, matrix operand order checked by reading, not run on hardware
//
#include <hip/hip_runtime.h>
#include <math.h>

// ---------------- types ----------------
typedef unsigned short u16;
typedef __attribute__((ext_vector_type(8)))  u16    u16x8;
typedef __attribute__((ext_vector_type(16))) __bf16 v16bf;
typedef __attribute__((ext_vector_type(8)))  float  v8f;

#define DEV static __device__ __forceinline__

// ---------------- problem dims ----------------
#define BB 64      // batch
#define TT 200     // seq length
#define TP 208     // padded seq length (multiple of 16)
#define EE 256     // embed
#define HH 256     // hidden
#define H3 768     // 3*H
#define AD 256     // attn dim
#define VV 20000   // vocab
#define MM (BB*TP) // 13312, multiple of 256

// ---------------- bf16 helpers ----------------
DEV u16 f2bf(float f){ union{float f;unsigned u;} v; v.f=f; unsigned u=v.u;
  u += 0x7fffu + ((u>>16)&1u); return (u16)(u>>16); }
DEV float bf2f(u16 h){ union{unsigned u;float f;} v; v.u=((unsigned)h)<<16; return v.f; }

// WMMA bf16 16x16x32 fragment loader (row-major, K contiguous).
// A-operand: caller passes row = m0 + (lane&15).  B-operand (W stored [N,K]
// row-major): caller passes row = n0 + (lane&15).  Per ISA layout, lane half
// selects K sub-block: lanes 0-15 hold K {0..7,16..23}, lanes 16-31 hold
// K {8..15,24..31} of the 32-wide K step.
DEV v16bf frag_ld(const u16* buf, int row, int ld, int k){
  const int lane = threadIdx.x & 31;
  const u16* p = buf + (size_t)row*ld + k + ((lane & 16) ? 8 : 0);
  union { u16x8 s[2]; v16bf v; } u;
  u.s[0] = *(const u16x8*)p;
  u.s[1] = *(const u16x8*)(p + 16);
  return u.v;
}

DEV v8f wmma_bf16(v16bf a, v16bf b, v8f c){
  return __builtin_amdgcn_wmma_f32_16x16x32_bf16(
      /*neg_a=*/false, a, /*neg_b=*/false, b,
      /*c_mod=*/(short)0, c, /*reuse_a=*/false, /*reuse_b=*/false);
}

// ---------------- tiny elementwise kernels ----------------
__global__ void k_cvt_bf16(const float* __restrict__ s, u16* __restrict__ d, int n){
  for (int i = blockIdx.x*blockDim.x + threadIdx.x; i < n; i += gridDim.x*blockDim.x)
    d[i] = f2bf(s[i]);
}

// xe[b][t][e] (bf16, padded to TP rows; pad rows = 0)
__global__ void k_gather(const int* __restrict__ x, const float* __restrict__ emb,
                         u16* __restrict__ xe){
  const int n = MM*EE;
  for (int i = blockIdx.x*blockDim.x + threadIdx.x; i < n; i += gridDim.x*blockDim.x){
    int e = i % EE, row = i / EE, t = row % TP, b = row / TP;
    float v = 0.f;
    if (t < TT){ int id = x[b*TT + t]; v = emb[(size_t)id*EE + e]; }
    xe[i] = f2bf(v);
  }
}

// aT[b][d][s] = a[b][s][d]
__global__ void k_transpose(const u16* __restrict__ a, u16* __restrict__ at){
  const int n = BB*AD*TP;
  for (int i = blockIdx.x*blockDim.x + threadIdx.x; i < n; i += gridDim.x*blockDim.x){
    int s = i % TP, rd = i / TP, d = rd % AD, b = rd / AD;
    at[i] = a[((size_t)b*TP + s)*AD + d];
  }
}

// ---------------- generic WMMA GEMM ----------------
// C[M,N] = A[M,K]·W[N,K]^T (+ A2·W2^T) + bias.  A,W bf16 row-major.
// Block = 8 waves; wave w owns 32 rows (2 m-tiles, B-frag reused 2x),
// block owns 64 cols.  M must be a multiple of 256 (gridDim.y = M/256);
// N a multiple of 16.  Out-of-range N tiles are clamped to N-16 so the
// K-loop is branch-free; only stores are guarded.
// outBf16: store bf16; remap: rows are (b*TP + t), store only t<TT to
// compact [b*TT + t] rows of the fp32 output.
__global__ void __launch_bounds__(256,1)
k_gemm(const u16* __restrict__ A, int lda,
       const u16* __restrict__ W, int ldb,
       const u16* __restrict__ A2, int lda2,
       const u16* __restrict__ W2, int ldb2,
       const float* __restrict__ bias,
       void* __restrict__ Cout, int ldc,
       int N, int K, int outBf16, int remap){
  const int lane = threadIdx.x & 31;
  const int wv   = threadIdx.x >> 5;
  const int n0   = blockIdx.x * 64;
  const int m0   = blockIdx.y * 256 + wv*32;
  bool nv[4]; int ncol[4];
#pragma unroll
  for (int nt=0; nt<4; ++nt){
    int c = n0 + nt*16;
    nv[nt]   = c < N;
    ncol[nt] = (nv[nt] ? c : (N - 16)) + (lane & 15);   // clamped: always in-bounds
  }
  v8f acc[8] = {};
  const int ar0 = m0 +      (lane & 15);
  const int ar1 = m0 + 16 + (lane & 15);

  for (int k=0; k<K; k+=32){
    v16bf a0 = frag_ld(A, ar0, lda, k);
    v16bf a1 = frag_ld(A, ar1, lda, k);
#pragma unroll
    for (int nt=0; nt<4; ++nt){
      const u16* wp = W + (size_t)ncol[nt]*ldb + k;
      __builtin_prefetch((const void*)(wp + 32), 0, 1);
      v16bf b = frag_ld(W, ncol[nt], ldb, k);
      acc[nt]   = wmma_bf16(a0, b, acc[nt]);
      acc[4+nt] = wmma_bf16(a1, b, acc[4+nt]);
    }
  }
  if (A2){
    for (int k=0; k<K; k+=32){
      v16bf a0 = frag_ld(A2, ar0, lda2, k);
      v16bf a1 = frag_ld(A2, ar1, lda2, k);
#pragma unroll
      for (int nt=0; nt<4; ++nt){
        v16bf b = frag_ld(W2, ncol[nt], ldb2, k);
        acc[nt]   = wmma_bf16(a0, b, acc[nt]);
        acc[4+nt] = wmma_bf16(a1, b, acc[4+nt]);
      }
    }
  }

  const int rofs = (lane>=16)?8:0;
#pragma unroll
  for (int half=0; half<2; ++half){
    const int rbase = m0 + half*16 + rofs;
#pragma unroll
    for (int nt=0; nt<4; ++nt){
      if (!nv[nt]) continue;
      int col = n0 + nt*16 + (lane&15);
      float bv = bias ? bias[col] : 0.f;
#pragma unroll
      for (int r=0; r<8; ++r){
        int row = rbase + r;
        float v = acc[half*4+nt][r] + bv;
        if (remap){
          int t = row % TP;
          if (t < TT)
            ((float*)Cout)[((size_t)(row/TP)*TT + t)*ldc + col] = v;
        } else if (outBf16){
          ((u16*)Cout)[(size_t)row*ldc + col] = f2bf(v);
        } else {
          ((float*)Cout)[(size_t)row*ldc + col] = v;
        }
      }
    }
  }
}

// ---------------- GRU recurrence (persistent single workgroup) ----------------
// h kept bf16-resident in LDS; each step: Gh = h·W_hh^T via WMMA across 16
// waves (4 (m-tile, h-col-tile) triples each, 3 gate tiles per triple, 8
// K-steps => 96 wmma/wave/step), then per-element gate math + h update.
__global__ void __launch_bounds__(512,1)
k_gru(const float* __restrict__ Gi, const u16* __restrict__ Whh,
      const float* __restrict__ bhh, const int* __restrict__ xlen,
      u16* __restrict__ hs){
  __shared__ u16 hsm[BB*HH];            // 32 KB, h in bf16
  const int tid = threadIdx.x, lane = tid & 31, wv = tid >> 5;
  for (int i=tid; i<BB*HH; i+=512) hsm[i]=0;
  __syncthreads();

  for (int t=0; t<TT; ++t){
    v8f aR[4]={}, aZ[4]={}, aN[4]={};
#pragma unroll
    for (int i=0;i<4;++i){
      int p = wv*4 + i, mt = p>>4, jt = p&15;
      int arow = mt*16 + (lane&15);
      int wrow = jt*16 + (lane&15);
      for (int k=0; k<HH; k+=32){
        v16bf a  = frag_ld(hsm, arow,        HH, k);
        v16bf bR = frag_ld(Whh, wrow,        HH, k);
        v16bf bZ = frag_ld(Whh, HH   + wrow, HH, k);
        v16bf bN = frag_ld(Whh, 2*HH + wrow, HH, k);
        aR[i] = wmma_bf16(a, bR, aR[i]);
        aZ[i] = wmma_bf16(a, bZ, aZ[i]);
        aN[i] = wmma_bf16(a, bN, aN[i]);
      }
    }
    __syncthreads();                     // all reads of h done
#pragma unroll
    for (int i=0;i<4;++i){
      int p = wv*4 + i, mt = p>>4, jt = p&15;
      int jl = lane & 15, mofs = (lane>=16)?8:0;
#pragma unroll
      for (int r=0;r<8;++r){
        int m = mt*16 + mofs + r;        // batch row (unique per thread/elem)
        int j = jt*16 + jl;              // hidden col
        size_t gib = ((size_t)m*TP + t)*H3;
        float gir = Gi[gib + j], giz = Gi[gib + HH + j], gin = Gi[gib + 2*HH + j];
        float ghr = aR[i][r] + bhh[j];
        float ghz = aZ[i][r] + bhh[HH + j];
        float ghn = aN[i][r] + bhh[2*HH + j];
        float rg = 1.f/(1.f+__expf(-(gir+ghr)));
        float zg = 1.f/(1.f+__expf(-(giz+ghz)));
        float ng = tanhf(gin + rg*ghn);
        float hold = bf2f(hsm[m*HH + j]);
        float hnew = (1.f-zg)*ng + zg*hold;
        hsm[m*HH + j] = f2bf(hnew);
        hs[((size_t)m*TP + t)*HH + j] = (t < xlen[m]) ? f2bf(hnew) : (u16)0;
      }
    }
    __syncthreads();                     // h fully updated for next step
  }
  // zero pad rows t in [TT,TP)
  for (int i=tid; i < BB*(TP-TT)*HH; i+=512){
    int j = i % HH, r = i / HH, tt = TT + (r % (TP-TT)), b = r/(TP-TT);
    hs[((size_t)b*TP + tt)*HH + j] = 0;
  }
}

// ---------------- causal self-attention over a[b][t][d] ----------------
// Block = (query tile qt, batch b), 4 waves. Phase1: S = a_q·a_k^T /AD with
// causal mask (key tiles split across waves). Phase2: row softmax in LDS,
// probs -> bf16. Phase3: ctx = P·a_k via WMMA using pre-transposed aT.
__global__ void __launch_bounds__(128,1)
k_attn(const u16* __restrict__ a, const u16* __restrict__ at,
       u16* __restrict__ ctx){
  __shared__ float sS[16*224];           // scores (14 KB)
  __shared__ u16   sP[16*224];           // probs bf16 (7 KB)
  const int qt = blockIdx.x, b = blockIdx.y;
  const int tid = threadIdx.x, lane = tid & 31, wv = tid >> 5;
  const int nk = qt + 1;                 // causal: key tiles 0..qt
  const int q0 = qt*16;

  // phase 1: score tiles
  for (int kt = wv; kt < nk; kt += 4){
    v8f s = {};
    int arow = b*TP + q0 + (lane&15);
    int brow = b*TP + kt*16 + (lane&15);
    for (int k=0; k<AD; k+=32){
      v16bf aq = frag_ld(a, arow, AD, k);
      v16bf ak = frag_ld(a, brow, AD, k);
      s = wmma_bf16(aq, ak, s);
    }
    int sl = lane & 15, rofs = (lane>=16)?8:0;
#pragma unroll
    for (int r=0;r<8;++r){
      int q = q0 + rofs + r, sg = kt*16 + sl;
      sS[(rofs+r)*224 + kt*16 + sl] = (sg <= q) ? s[r]*(1.f/AD) : -1e30f;
    }
  }
  __syncthreads();

  // phase 2: softmax per query row (serial; <=208 cols)
  if (tid < 16){
    int ncol = nk*16;
    float* row = &sS[tid*224];
    float mx = -1e30f;
    for (int c=0;c<ncol;++c) mx = fmaxf(mx, row[c]);
    float sum = 0.f;
    for (int c=0;c<ncol;++c){ float e = __expf(row[c]-mx); row[c]=e; sum+=e; }
    float inv = 1.f/sum;
    u16* pr = &sP[tid*224];
    for (int c=0;c<ncol;++c)  pr[c] = f2bf(row[c]*inv);
    for (int c=ncol;c<224;++c) pr[c] = 0;   // zero-pad for 32-wide K steps
  }
  __syncthreads();

  // phase 3: ctx = P @ a_k ; wave wv owns 64 of the 256 output dims
  v8f c4[4] = {};
  const int d0 = wv*64;
  const int kcn = (nk*16 + 31)/32;
  for (int kc=0; kc<kcn; ++kc){
    v16bf p = frag_ld(sP, (lane&15), 224, kc*32);
#pragma unroll
    for (int nt=0; nt<4; ++nt){
      v16bf bv = frag_ld(at, b*AD + d0 + nt*16 + (lane&15), TP, kc*32);
      c4[nt] = wmma_bf16(p, bv, c4[nt]);
    }
  }
  int rofs = (lane>=16)?8:0;
#pragma unroll
  for (int nt=0; nt<4; ++nt){
    int d = d0 + nt*16 + (lane&15);
#pragma unroll
    for (int r=0;r<8;++r){
      int q = q0 + rofs + r;
      ctx[((size_t)b*TP + q)*AD + d] = f2bf(c4[nt][r]);
    }
  }
}

// ---------------- host launcher ----------------
extern "C" void kernel_launch(void* const* d_in, const int* in_sizes, int n_in,
                              void* d_out, int out_size, void* d_ws, size_t ws_size,
                              hipStream_t stream){
  (void)in_sizes; (void)n_in; (void)out_size; (void)ws_size;
  const int*   x    = (const int*)  d_in[0];
  const int*   xlen = (const int*)  d_in[1];
  const float* emb  = (const float*)d_in[2];
  const float* Wih  = (const float*)d_in[3];
  const float* Whh  = (const float*)d_in[4];
  const float* bih  = (const float*)d_in[5];
  const float* bhh  = (const float*)d_in[6];
  const float* Wat  = (const float*)d_in[7];
  const float* bat  = (const float*)d_in[8];
  const float* Wo   = (const float*)d_in[9];
  const float* bo   = (const float*)d_in[10];
  float* out = (float*)d_out;

  char* ws = (char*)d_ws;
  size_t off = 0;
  auto carve = [&](size_t bytes)->char*{
    char* p = ws + off; off += (bytes + 255) & ~(size_t)255; return p; };
  u16*   xe   = (u16*)  carve((size_t)MM*EE*2);
  float* Gi   = (float*)carve((size_t)MM*H3*4);
  u16*   hsb  = (u16*)  carve((size_t)MM*HH*2);
  u16*   ab   = (u16*)  carve((size_t)MM*AD*2);
  u16*   atb  = (u16*)  carve((size_t)BB*AD*TP*2 + 256);  // slack for K-pad reads
  u16*   ctxb = (u16*)  carve((size_t)MM*AD*2);
  u16*   Wihb = (u16*)  carve((size_t)H3*EE*2);
  u16*   Whhb = (u16*)  carve((size_t)H3*HH*2);
  u16*   Watb = (u16*)  carve((size_t)AD*HH*2);
  u16*   Wob  = (u16*)  carve((size_t)VV*(AD+HH)*2);

  // weight / activation conversion + embedding gather
  k_cvt_bf16<<<256, 256, 0, stream>>>(Wih, Wihb, H3*EE);
  k_cvt_bf16<<<256, 256, 0, stream>>>(Whh, Whhb, H3*HH);
  k_cvt_bf16<<<64,  256, 0, stream>>>(Wat, Watb, AD*HH);
  k_cvt_bf16<<<2048,256, 0, stream>>>(Wo,  Wob,  VV*(AD+HH));
  k_gather  <<<2048,256, 0, stream>>>(x, emb, xe);

  // Gi = xe @ W_ih^T + b_ih  (fp32, [B,TP,3H])
  k_gemm<<<dim3(H3/64, MM/256), 256, 0, stream>>>(
      xe, EE, Wihb, EE, nullptr,0, nullptr,0, bih, Gi, H3, H3, EE, 0, 0);

  // sequential GRU scan, h resident in LDS
  k_gru<<<1, 512, 0, stream>>>(Gi, Whhb, bhh, xlen, hsb);

  // a = hs @ W_attn^T + b_attn  (bf16 out)
  k_gemm<<<dim3(AD/64, MM/256), 256, 0, stream>>>(
      hsb, HH, Watb, HH, nullptr,0, nullptr,0, bat, ab, AD, AD, HH, 1, 0);

  // aT + causal attention -> ctx (bf16)
  k_transpose<<<1024, 256, 0, stream>>>(ab, atb);
  k_attn<<<dim3(TP/16, BB), 128, 0, stream>>>(ab, atb, ctxb);

  // out = ctx @ Wo[:, :A]^T + hs @ Wo[:, A:]^T + b_out  (fp32, rows remapped TP->TT)
  k_gemm<<<dim3((VV+63)/64, MM/256), 256, 0, stream>>>(
      ctxb, AD, Wob, AD+HH, hsb, HH, Wob + AD, AD+HH, bo, out, VV, VV, AD, 0, 1);
}
